// SE3TransformerTrIP_36172214567115
// MI455X (gfx1250) — matrix operand order repaired and checked by reading, not compiled, not run on hardware
//
#include <hip/hip_runtime.h>
#include <math.h>

#define NN 10000     // nodes
#define EE 100000    // edges
#define CCH 16       // channels
#define KVC 8        // KV channels
#define HH 4         // heads
#define NLAYERS 2
#define MIDW 32

typedef __attribute__((ext_vector_type(16))) _Float16 v16h;
typedef __attribute__((ext_vector_type(8)))  float    v8f;

#define WAVES_PER_BLOCK 4
#define TPB (WAVES_PER_BLOCK * 32)

// ---------------------------------------------------------------- WMMA helpers

__device__ __forceinline__ v8f wmma_16x16x32(v16h a, v16h b, v8f c) {
  // D(16x16 f32) = A(16x32 f16) * B(32x16 f16) + C, f32 accumulate
  return __builtin_amdgcn_wmma_f32_16x16x32_f16(false, a, false, b, (short)0, c,
                                                false, false);
}

// A-matrix fragment from a row-major 16x32 f16 LDS tile.
// Layout (16-bit A 16x32, matching packed B): lanes 0-15 -> M=lane, K 0-15;
// lanes 16-31 -> M=lane-16, K 16-31.
__device__ __forceinline__ v16h load_a_frag(const _Float16* sA, int lane) {
  const int m  = lane & 15;
  const int kb = (lane >> 4) << 4;   // 0 or 16
  v16h a;
#pragma unroll
  for (int t = 0; t < 8; ++t) {
    a[t]     = sA[m * 32 + kb + t];
    a[t + 8] = sA[m * 32 + kb + 8 + t];
  }
  return a;
}

// B fragment pre-packed in global memory: 16 contiguous halves per lane.
__device__ __forceinline__ v16h load_b_frag(const _Float16* g) {
  v16h b;
#pragma unroll
  for (int t = 0; t < 16; ++t) b[t] = g[t];
  return b;
}

// ---------------------------------------------------------------- small utils

__device__ __forceinline__ void atomicMaxF(float* addr, float val) {
  unsigned int* a = (unsigned int*)addr;
  unsigned int old = *a;
  while (__uint_as_float(old) < val) {
    unsigned int assumed = old;
    old = atomicCAS(a, assumed, __float_as_uint(val));
    if (old == assumed) break;
  }
}

__global__ void fill_kernel(float* p, float v, int n) {
  int i = blockIdx.x * blockDim.x + threadIdx.x;
  if (i < n) p[i] = v;
}

// r = |rel|, u = rel/r, edge0 = r
__global__ void basis_kernel(const float* rel, float* u, float* edge0, int E) {
  int e = blockIdx.x * blockDim.x + threadIdx.x;
  if (e >= E) return;
  float x = rel[e * 3 + 0], y = rel[e * 3 + 1], z = rel[e * 3 + 2];
  float r = sqrtf(x * x + y * y + z * z);
  float inv = 1.f / r;
  u[e * 3 + 0] = x * inv;
  u[e * 3 + 1] = y * inv;
  u[e * 3 + 2] = z * inv;
  edge0[e] = r;
}

// Convert w2 (32x32) and w3 (32xnout) f32 -> packed f16 B fragments.
// tile 0,1 -> w2; tile >=2 -> w3 tile (tile-2). One wave per tile.
// Lane layout of a 32x16 B tile: N = lane&15, K = t + 16*(lane>=16), t=0..15.
__global__ void prep_kernel(const float* w2, const float* w3, int nout,
                            _Float16* w2f, _Float16* w3f) {
  int tile = blockIdx.x;
  int lane = threadIdx.x;
  const float* W;
  _Float16* D;
  int ncols, nt;
  if (tile < 2) { W = w2; D = w2f; ncols = 32;   nt = tile; }
  else          { W = w3; D = w3f; ncols = nout; nt = tile - 2; }
  int n = lane & 15;
  int koff = (lane >> 4) * 16;
#pragma unroll
  for (int t = 0; t < 16; ++t)
    D[((size_t)nt * 32 + lane) * 16 + t] =
        (_Float16)W[(size_t)(t + koff) * ncols + nt * 16 + n];
}

// ---------------------------------------------------------------- WMMA kv conv

struct PairP {
  const float *w1, *b1, *g1, *bb1, *b2, *g2, *bb2;
  const _Float16 *w2f, *w3f;
  int nt;   // nout/16
};

struct KvArgs {
  const float* edge0;
  const int* src;
  const float* u;
  const float* scale;
  const float* f0;   // (N,16)
  const float* f1;   // (N,16,3)
  float* kv0;        // (E,16)
  float* kv1;        // (E,16,3)
  int ntile;
  PairP p[4];        // order: 00, 01, 10, 11
};

struct KvShared {
  _Float16 sA[16 * 32];   // A tile (h1 or h2), f16 row-major
  float sX[16 * 32];      // f32 scratch (GEMM dumps)
  float f0[16 * 16];
  float f1[16 * 16 * 3];
  float f1u[16 * 16];
  float uu[16 * 3];
  float sc[16];
  float x0[16];
  float kv0[16 * 16];
  float kv1[16 * 16 * 3];
};

__global__ __launch_bounds__(TPB) void kv_kernel(KvArgs A) {
  __shared__ KvShared sh[WAVES_PER_BLOCK];
  const int wave = threadIdx.x >> 5;
  const int lane = threadIdx.x & 31;
  const int tile = blockIdx.x * WAVES_PER_BLOCK + wave;
  const bool valid = tile < A.ntile;
  const int t0 = valid ? tile : 0;
  KvShared& S = sh[wave];

  // stage per-edge data
  for (int i = lane; i < 16; i += 32) {
    int e = t0 * 16 + i;
    S.sc[i] = A.scale[e];
    S.x0[i] = A.edge0[e];
    S.uu[i * 3 + 0] = A.u[e * 3 + 0];
    S.uu[i * 3 + 1] = A.u[e * 3 + 1];
    S.uu[i * 3 + 2] = A.u[e * 3 + 2];
  }
  for (int i = lane; i < 256; i += 32) {
    int m = i >> 4, c = i & 15;
    int e = t0 * 16 + m;
    int s = A.src[e];
    S.f0[i] = A.f0[s * 16 + c];
    S.f1[i * 3 + 0] = A.f1[s * 48 + c * 3 + 0];
    S.f1[i * 3 + 1] = A.f1[s * 48 + c * 3 + 1];
    S.f1[i * 3 + 2] = A.f1[s * 48 + c * 3 + 2];
  }
  for (int i = lane; i < 256; i += 32) S.kv0[i] = 0.f;
  for (int i = lane; i < 768; i += 32) S.kv1[i] = 0.f;
  __syncthreads();
  for (int i = lane; i < 256; i += 32) {
    int m = i >> 4;
    S.f1u[i] = S.f1[i * 3 + 0] * S.uu[m * 3 + 0] +
               S.f1[i * 3 + 1] * S.uu[m * 3 + 1] +
               S.f1[i * 3 + 2] * S.uu[m * 3 + 2];
  }
  __syncthreads();

  for (int pp = 0; pp < 4; ++pp) {
    const PairP P = A.p[pp];

    // h1 = relu(LN(x0*w1 + b1)); one row per lane (lanes 0..15)
    if (lane < 16) {
      const int m = lane;
      const float x = S.x0[m];
      float v[32];
      float mu = 0.f;
#pragma unroll
      for (int k = 0; k < 32; ++k) { v[k] = x * P.w1[k] + P.b1[k]; mu += v[k]; }
      mu *= (1.f / 32.f);
      float var = 0.f;
#pragma unroll
      for (int k = 0; k < 32; ++k) { float d = v[k] - mu; var += d * d; }
      var *= (1.f / 32.f);
      const float rs = rsqrtf(var + 1e-5f);
#pragma unroll
      for (int k = 0; k < 32; ++k) {
        float h = (v[k] - mu) * rs * P.g1[k] + P.bb1[k];
        S.sA[m * 32 + k] = (_Float16)fmaxf(h, 0.f);
      }
    }
    __syncthreads();

    // GEMM1: (16x32)*(32x32) via 2 WMMAs, dump pre-h2 f32 to sX
    {
      const v16h afrag = load_a_frag(S.sA, lane);
#pragma unroll
      for (int nt2 = 0; nt2 < 2; ++nt2) {
        const v16h bfrag = load_b_frag(P.w2f + ((size_t)nt2 * 32 + lane) * 16);
        v8f acc = {};
        acc = wmma_16x16x32(afrag, bfrag, acc);
        const int n = lane & 15;
        const int mr = (lane >> 4) * 8;
#pragma unroll
        for (int r = 0; r < 8; ++r) S.sX[(mr + r) * 32 + nt2 * 16 + n] = acc[r];
      }
    }
    __syncthreads();

    // h2 = relu(LN(pre + b2))
    if (lane < 16) {
      const int m = lane;
      float v[32];
      float mu = 0.f;
#pragma unroll
      for (int k = 0; k < 32; ++k) { v[k] = S.sX[m * 32 + k] + P.b2[k]; mu += v[k]; }
      mu *= (1.f / 32.f);
      float var = 0.f;
#pragma unroll
      for (int k = 0; k < 32; ++k) { float d = v[k] - mu; var += d * d; }
      var *= (1.f / 32.f);
      const float rs = rsqrtf(var + 1e-5f);
#pragma unroll
      for (int k = 0; k < 32; ++k) {
        float h = (v[k] - mu) * rs * P.g2[k] + P.bb2[k];
        S.sA[m * 32 + k] = (_Float16)fmaxf(h, 0.f);
      }
    }
    __syncthreads();

    // GEMM2: W tiles = h2 @ w3, contract with basis immediately.
    const v16h a2 = load_a_frag(S.sA, lane);
    for (int tt = 0; tt < P.nt; ++tt) {
      const v16h bfrag = load_b_frag(P.w3f + ((size_t)tt * 32 + lane) * 16);
      v8f acc = {};
      acc = wmma_16x16x32(a2, bfrag, acc);
      const int n = lane & 15;
      const int mr = (lane >> 4) * 8;
#pragma unroll
      for (int r = 0; r < 8; ++r) S.sX[(mr + r) * 16 + n] = acc[r];
      __syncthreads();
      if (lane < 16) {
        const int m = lane;
        const float sc = S.sc[m];
        if (pp == 0) {                      // '00': kv0[d=tt] += s * W.f0
          float s = 0.f;
#pragma unroll
          for (int cc = 0; cc < 16; ++cc) s += S.sX[m * 16 + cc] * S.f0[m * 16 + cc];
          S.kv0[m * 16 + tt] += sc * s;
        } else if (pp == 2) {               // '10': kv0[d=tt] += s * W.(u·f1)
          float s = 0.f;
#pragma unroll
          for (int cc = 0; cc < 16; ++cc) s += S.sX[m * 16 + cc] * S.f1u[m * 16 + cc];
          S.kv0[m * 16 + tt] += sc * s;
        } else if (pp == 1) {               // '01': kv1[d,o] += s*u_o * W.f0
          float s = 0.f;
#pragma unroll
          for (int cc = 0; cc < 16; ++cc) s += S.sX[m * 16 + cc] * S.f0[m * 16 + cc];
          s *= sc;
          S.kv1[(m * 16 + tt) * 3 + 0] += s * S.uu[m * 3 + 0];
          S.kv1[(m * 16 + tt) * 3 + 1] += s * S.uu[m * 3 + 1];
          S.kv1[(m * 16 + tt) * 3 + 2] += s * S.uu[m * 3 + 2];
        } else {                            // '11': j = tt>>4, d = tt&15
          const int j = tt >> 4, d = tt & 15;
          const float ux = S.uu[m * 3 + 0], uy = S.uu[m * 3 + 1], uz = S.uu[m * 3 + 2];
          float s0 = 0.f, s1 = 0.f, s2 = 0.f;
#pragma unroll
          for (int cc = 0; cc < 16; ++cc) {
            const float w  = S.sX[m * 16 + cc];
            const float fx = S.f1[(m * 16 + cc) * 3 + 0];
            const float fy = S.f1[(m * 16 + cc) * 3 + 1];
            const float fz = S.f1[(m * 16 + cc) * 3 + 2];
            if (j == 0) {                   // J0 = I
              s0 += w * fx; s1 += w * fy; s2 += w * fz;
            } else if (j == 1) {            // (J1 f)_o = (f x u)_o
              s0 += w * (fy * uz - fz * uy);
              s1 += w * (fz * ux - fx * uz);
              s2 += w * (fx * uy - fy * ux);
            } else {                        // (J2 f)_o = u_o(u.f) - f_o/3
              const float du = fx * ux + fy * uy + fz * uz;
              s0 += w * (ux * du - fx * (1.f / 3.f));
              s1 += w * (uy * du - fy * (1.f / 3.f));
              s2 += w * (uz * du - fz * (1.f / 3.f));
            }
          }
          S.kv1[(m * 16 + d) * 3 + 0] += sc * s0;
          S.kv1[(m * 16 + d) * 3 + 1] += sc * s1;
          S.kv1[(m * 16 + d) * 3 + 2] += sc * s2;
        }
      }
      __syncthreads();
    }
  }

  if (valid) {
    for (int i = lane; i < 256; i += 32)
      A.kv0[(size_t)(tile * 16 + (i >> 4)) * 16 + (i & 15)] = S.kv0[i];
    for (int i = lane; i < 768; i += 32) {
      int m = i / 48, rem = i % 48;
      A.kv1[(size_t)(tile * 16 + m) * 48 + rem] = S.kv1[i];
    }
  }
}

// ---------------------------------------------------------------- attention

__global__ void q_kernel(const float* f0, const float* f1, const float* q0,
                         const float* q1, float* qf, int N) {
  int n = blockIdx.x * blockDim.x + threadIdx.x;
  if (n >= N) return;
#pragma unroll
  for (int d = 0; d < 8; ++d) {
    float s = 0.f;
    for (int c = 0; c < 16; ++c) s += q0[d * 16 + c] * f0[n * 16 + c];
    qf[n * 32 + d] = s;
  }
#pragma unroll
  for (int d = 0; d < 8; ++d) {
    float s0 = 0.f, s1 = 0.f, s2 = 0.f;
    for (int c = 0; c < 16; ++c) {
      float w = q1[d * 16 + c];
      s0 += w * f1[n * 48 + c * 3 + 0];
      s1 += w * f1[n * 48 + c * 3 + 1];
      s2 += w * f1[n * 48 + c * 3 + 2];
    }
    qf[n * 32 + 8 + d * 3 + 0] = s0;
    qf[n * 32 + 8 + d * 3 + 1] = s1;
    qf[n * 32 + 8 + d * 3 + 2] = s2;
  }
}

__global__ void logits_kernel(const float* kv0, const float* kv1, const float* qf,
                              const int* dst, float* logits, float* mbuf, int E) {
  int e = blockIdx.x * blockDim.x + threadIdx.x;
  if (e >= E) return;
  float key[32];
#pragma unroll
  for (int d = 0; d < 8; ++d) key[d] = kv0[e * 16 + d];
#pragma unroll
  for (int d = 0; d < 8; ++d)
    for (int o = 0; o < 3; ++o) key[8 + d * 3 + o] = kv1[e * 48 + d * 3 + o];
  int n = dst[e];
  const float inv = 0.35355339059327373f;  // 1/sqrt(8)
#pragma unroll
  for (int h = 0; h < 4; ++h) {
    float s = 0.f;
    for (int t = 0; t < 8; ++t) s += key[h * 8 + t] * qf[n * 32 + h * 8 + t];
    s *= inv;
    logits[e * 4 + h] = s;
    atomicMaxF(&mbuf[n * 4 + h], s);
  }
}

__global__ void w_kernel(const float* scale, const int* dst, const float* mbuf,
                         float* wbuf, float* zbuf, int E) {
  int e = blockIdx.x * blockDim.x + threadIdx.x;
  if (e >= E) return;
  int n = dst[e];
  float sc = scale[e];
#pragma unroll
  for (int h = 0; h < 4; ++h) {
    float mm = mbuf[n * 4 + h];
    if (!(mm > -1e30f)) mm = 0.f;
    float w = sc * expf(wbuf[e * 4 + h] - mm);
    wbuf[e * 4 + h] = w;
    atomicAdd(&zbuf[n * 4 + h], w);
  }
}

__global__ void agg_kernel(const float* wbuf, const float* zbuf, const int* dst,
                           const float* kv0, const float* kv1, float* agg0,
                           float* agg1, int E) {
  int e = blockIdx.x * blockDim.x + threadIdx.x;
  if (e >= E) return;
  int n = dst[e];
  float al[4];
#pragma unroll
  for (int h = 0; h < 4; ++h) al[h] = wbuf[e * 4 + h] / (zbuf[n * 4 + h] + 1e-9f);
#pragma unroll
  for (int kk = 0; kk < 8; ++kk) {
    float a = al[kk >> 1];
    atomicAdd(&agg0[n * 8 + kk], a * kv0[e * 16 + 8 + kk]);
    atomicAdd(&agg1[(n * 8 + kk) * 3 + 0], a * kv1[e * 48 + (8 + kk) * 3 + 0]);
    atomicAdd(&agg1[(n * 8 + kk) * 3 + 1], a * kv1[e * 48 + (8 + kk) * 3 + 1]);
    atomicAdd(&agg1[(n * 8 + kk) * 3 + 2], a * kv1[e * 48 + (8 + kk) * 3 + 2]);
  }
}

__global__ void node_kernel(const float* agg0, const float* agg1, const float* f0in,
                            const float* f1in, const float* proj0, const float* proj1,
                            const float* n0g, const float* n0b, const float* n1g,
                            const float* n1b, float* f0out, float* f1out, int N) {
  int n = blockIdx.x * blockDim.x + threadIdx.x;
  if (n >= N) return;
  // degree 0
  float y0[16], nr[16];
  for (int d = 0; d < 16; ++d) {
    float s = 0.f;
    for (int c = 0; c < 8; ++c)  s += proj0[d * 24 + c] * agg0[n * 8 + c];
    for (int c = 0; c < 16; ++c) s += proj0[d * 24 + 8 + c] * f0in[n * 16 + c];
    y0[d] = s;
  }
  float mu = 0.f;
  for (int d = 0; d < 16; ++d) { nr[d] = sqrtf(y0[d] * y0[d] + 1e-12f); mu += nr[d]; }
  mu *= (1.f / 16.f);
  float var = 0.f;
  for (int d = 0; d < 16; ++d) { float x = nr[d] - mu; var += x * x; }
  var *= (1.f / 16.f);
  float rs = rsqrtf(var + 1e-5f);
  for (int d = 0; d < 16; ++d) {
    float ln = (nr[d] - mu) * rs * n0g[d] + n0b[d];
    f0out[n * 16 + d] = y0[d] * (ln / nr[d]);
  }
  // degree 1
  float y1[16][3];
  for (int d = 0; d < 16; ++d) {
    float s0 = 0.f, s1 = 0.f, s2 = 0.f;
    for (int c = 0; c < 8; ++c) {
      float w = proj1[d * 24 + c];
      s0 += w * agg1[(n * 8 + c) * 3 + 0];
      s1 += w * agg1[(n * 8 + c) * 3 + 1];
      s2 += w * agg1[(n * 8 + c) * 3 + 2];
    }
    for (int c = 0; c < 16; ++c) {
      float w = proj1[d * 24 + 8 + c];
      s0 += w * f1in[n * 48 + c * 3 + 0];
      s1 += w * f1in[n * 48 + c * 3 + 1];
      s2 += w * f1in[n * 48 + c * 3 + 2];
    }
    y1[d][0] = s0; y1[d][1] = s1; y1[d][2] = s2;
  }
  mu = 0.f;
  for (int d = 0; d < 16; ++d) {
    nr[d] = sqrtf(y1[d][0] * y1[d][0] + y1[d][1] * y1[d][1] + y1[d][2] * y1[d][2] + 1e-12f);
    mu += nr[d];
  }
  mu *= (1.f / 16.f);
  var = 0.f;
  for (int d = 0; d < 16; ++d) { float x = nr[d] - mu; var += x * x; }
  var *= (1.f / 16.f);
  rs = rsqrtf(var + 1e-5f);
  for (int d = 0; d < 16; ++d) {
    float ln = (nr[d] - mu) * rs * n1g[d] + n1b[d];
    float s = ln / nr[d];
    f1out[n * 48 + d * 3 + 0] = y1[d][0] * s;
    f1out[n * 48 + d * 3 + 1] = y1[d][1] * s;
    f1out[n * 48 + d * 3 + 2] = y1[d][2] * s;
  }
}

__global__ void edge0_kernel(float* edge0, const int* src, const float* f0, int E) {
  int e = blockIdx.x * blockDim.x + threadIdx.x;
  if (e >= E) return;
  edge0[e] += f0[src[e] * 16 + 0];
}

// ---------------------------------------------------------------- final conv

struct FinArgs {
  const float* edge0;
  const int* src;
  const int* dst;
  const float* u;
  const float* scale;
  const float* f0;
  const float* f1;
  float* out;
  int ntile;
  PairP p[2];  // 00, 10
};

struct FinShared {
  _Float16 sA[16 * 32];
  float sX[16 * 32];
  float f0[256];
  float f1u[256];
  float sc[16];
  float x0[16];
  float res[16];
};

__global__ __launch_bounds__(TPB) void final_kernel(FinArgs A) {
  __shared__ FinShared sh[WAVES_PER_BLOCK];
  const int wave = threadIdx.x >> 5;
  const int lane = threadIdx.x & 31;
  const int tile = blockIdx.x * WAVES_PER_BLOCK + wave;
  const bool valid = tile < A.ntile;
  const int t0 = valid ? tile : 0;
  FinShared& S = sh[wave];

  for (int i = lane; i < 16; i += 32) {
    int e = t0 * 16 + i;
    S.sc[i] = A.scale[e];
    S.x0[i] = A.edge0[e];
    S.res[i] = 0.f;
  }
  for (int i = lane; i < 256; i += 32) {
    int m = i >> 4, c = i & 15;
    int e = t0 * 16 + m;
    int s = A.src[e];
    S.f0[i] = A.f0[s * 16 + c];
    S.f1u[i] = A.f1[s * 48 + c * 3 + 0] * A.u[e * 3 + 0] +
               A.f1[s * 48 + c * 3 + 1] * A.u[e * 3 + 1] +
               A.f1[s * 48 + c * 3 + 2] * A.u[e * 3 + 2];
  }
  __syncthreads();

  for (int pp = 0; pp < 2; ++pp) {
    const PairP P = A.p[pp];
    if (lane < 16) {
      const int m = lane;
      const float x = S.x0[m];
      float v[32];
      float mu = 0.f;
#pragma unroll
      for (int k = 0; k < 32; ++k) { v[k] = x * P.w1[k] + P.b1[k]; mu += v[k]; }
      mu *= (1.f / 32.f);
      float var = 0.f;
#pragma unroll
      for (int k = 0; k < 32; ++k) { float d = v[k] - mu; var += d * d; }
      var *= (1.f / 32.f);
      const float rs = rsqrtf(var + 1e-5f);
#pragma unroll
      for (int k = 0; k < 32; ++k) {
        float h = (v[k] - mu) * rs * P.g1[k] + P.bb1[k];
        S.sA[m * 32 + k] = (_Float16)fmaxf(h, 0.f);
      }
    }
    __syncthreads();
    {
      const v16h afrag = load_a_frag(S.sA, lane);
#pragma unroll
      for (int nt2 = 0; nt2 < 2; ++nt2) {
        const v16h bfrag = load_b_frag(P.w2f + ((size_t)nt2 * 32 + lane) * 16);
        v8f acc = {};
        acc = wmma_16x16x32(afrag, bfrag, acc);
        const int n = lane & 15;
        const int mr = (lane >> 4) * 8;
#pragma unroll
        for (int r = 0; r < 8; ++r) S.sX[(mr + r) * 32 + nt2 * 16 + n] = acc[r];
      }
    }
    __syncthreads();
    if (lane < 16) {
      const int m = lane;
      float v[32];
      float mu = 0.f;
#pragma unroll
      for (int k = 0; k < 32; ++k) { v[k] = S.sX[m * 32 + k] + P.b2[k]; mu += v[k]; }
      mu *= (1.f / 32.f);
      float var = 0.f;
#pragma unroll
      for (int k = 0; k < 32; ++k) { float d = v[k] - mu; var += d * d; }
      var *= (1.f / 32.f);
      const float rs = rsqrtf(var + 1e-5f);
#pragma unroll
      for (int k = 0; k < 32; ++k) {
        float h = (v[k] - mu) * rs * P.g2[k] + P.bb2[k];
        S.sA[m * 32 + k] = (_Float16)fmaxf(h, 0.f);
      }
    }
    __syncthreads();
    {
      const v16h a2 = load_a_frag(S.sA, lane);
      const v16h bfrag = load_b_frag(P.w3f + (size_t)lane * 16);
      v8f acc = {};
      acc = wmma_16x16x32(a2, bfrag, acc);
      const int n = lane & 15;
      const int mr = (lane >> 4) * 8;
#pragma unroll
      for (int r = 0; r < 8; ++r) S.sX[(mr + r) * 16 + n] = acc[r];
    }
    __syncthreads();
    if (lane < 16) {
      const int m = lane;
      const float* f = (pp == 0) ? &S.f0[m * 16] : &S.f1u[m * 16];
      float s = 0.f;
#pragma unroll
      for (int cc = 0; cc < 16; ++cc) s += S.sX[m * 16 + cc] * f[cc];
      S.res[m] += S.sc[m] * s;
    }
    __syncthreads();
  }
  if (valid && lane < 16) {
    int e = tile * 16 + lane;
    atomicAdd(&A.out[A.dst[e]], S.res[lane]);
  }
}

// ---------------------------------------------------------------- launch

extern "C" void kernel_launch(void* const* d_in, const int* in_sizes, int n_in,
                              void* d_out, int out_size, void* d_ws, size_t ws_size,
                              hipStream_t stream) {
  (void)in_sizes; (void)n_in; (void)out_size; (void)ws_size;
  const int E = EE, N = NN;

  const float* node0 = (const float*)d_in[0];
  const float* node1 = (const float*)d_in[1];
  const float* rel   = (const float*)d_in[2];
  const float* scale = (const float*)d_in[3];
  const int*   src   = (const int*)d_in[4];
  const int*   dst   = (const int*)d_in[5];

  struct Radial { const float *b1, *b2, *bb1, *bb2, *g1, *g2, *w1, *w2, *w3; };
  int idx = 6;
  auto take = [&]() -> Radial {
    Radial r;
    r.b1  = (const float*)d_in[idx++]; r.b2  = (const float*)d_in[idx++];
    r.bb1 = (const float*)d_in[idx++]; r.bb2 = (const float*)d_in[idx++];
    r.g1  = (const float*)d_in[idx++]; r.g2  = (const float*)d_in[idx++];
    r.w1  = (const float*)d_in[idx++]; r.w2  = (const float*)d_in[idx++];
    r.w3  = (const float*)d_in[idx++];
    return r;
  };
  // pytree order: params['final']{'00','10'}, then params['layers'][l]
  Radial fin00 = take();
  Radial fin10 = take();
  struct Layer {
    Radial kv[4];
    const float *n0b, *n0g, *n1b, *n1g, *proj0, *proj1, *q0, *q1;
  } L[NLAYERS];
  for (int l = 0; l < NLAYERS; ++l) {
    for (int p = 0; p < 4; ++p) L[l].kv[p] = take();   // '00','01','10','11'
    L[l].n0b = (const float*)d_in[idx++];
    L[l].n0g = (const float*)d_in[idx++];
    L[l].n1b = (const float*)d_in[idx++];
    L[l].n1g = (const float*)d_in[idx++];
    L[l].proj0 = (const float*)d_in[idx++];
    L[l].proj1 = (const float*)d_in[idx++];
    L[l].q0 = (const float*)d_in[idx++];
    L[l].q1 = (const float*)d_in[idx++];
  }

  // workspace carve-up
  size_t off = 0;
  auto alloc = [&](size_t bytes) -> char* {
    size_t o = (off + 255) & ~(size_t)255;
    off = o + bytes;
    return (char*)d_ws + o;
  };
  float* ubuf  = (float*)alloc(sizeof(float) * (size_t)E * 3);
  float* edge0 = (float*)alloc(sizeof(float) * (size_t)E);
  float* kv0   = (float*)alloc(sizeof(float) * (size_t)E * 16);
  float* kv1   = (float*)alloc(sizeof(float) * (size_t)E * 48);
  float* qf    = (float*)alloc(sizeof(float) * (size_t)N * 32);
  float* wlog  = (float*)alloc(sizeof(float) * (size_t)E * 4);
  float* mbuf  = (float*)alloc(sizeof(float) * (size_t)N * 4);
  float* zbuf  = (float*)alloc(sizeof(float) * (size_t)N * 4);
  float* agg0  = (float*)alloc(sizeof(float) * (size_t)N * 8);
  float* agg1  = (float*)alloc(sizeof(float) * (size_t)N * 24);
  float* f0a   = (float*)alloc(sizeof(float) * (size_t)N * 16);
  float* f1a   = (float*)alloc(sizeof(float) * (size_t)N * 48);
  float* f0b   = (float*)alloc(sizeof(float) * (size_t)N * 16);
  float* f1b   = (float*)alloc(sizeof(float) * (size_t)N * 48);

  const int nouts[4] = {256, 256, 256, 768};  // '00','01','10','11'
  _Float16 *w2f_l[NLAYERS][4], *w3f_l[NLAYERS][4];
  for (int l = 0; l < NLAYERS; ++l)
    for (int p = 0; p < 4; ++p) {
      w2f_l[l][p] = (_Float16*)alloc(sizeof(_Float16) * 2 * 32 * 16);
      w3f_l[l][p] = (_Float16*)alloc(sizeof(_Float16) * (size_t)(nouts[p] / 16) * 32 * 16);
    }
  _Float16 *w2f_fin[2], *w3f_fin[2];
  for (int p = 0; p < 2; ++p) {
    w2f_fin[p] = (_Float16*)alloc(sizeof(_Float16) * 2 * 32 * 16);
    w3f_fin[p] = (_Float16*)alloc(sizeof(_Float16) * 1 * 32 * 16);
  }

  // weight fragment conversion (f32 -> packed f16 B fragments)
  for (int l = 0; l < NLAYERS; ++l)
    for (int p = 0; p < 4; ++p)
      prep_kernel<<<dim3(2 + nouts[p] / 16), dim3(32), 0, stream>>>(
          L[l].kv[p].w2, L[l].kv[p].w3, nouts[p], w2f_l[l][p], w3f_l[l][p]);
  prep_kernel<<<dim3(3), dim3(32), 0, stream>>>(fin00.w2, fin00.w3, 16, w2f_fin[0], w3f_fin[0]);
  prep_kernel<<<dim3(3), dim3(32), 0, stream>>>(fin10.w2, fin10.w3, 16, w2f_fin[1], w3f_fin[1]);

  basis_kernel<<<(E + 255) / 256, 256, 0, stream>>>(rel, ubuf, edge0, E);

  const float* f0cur = node0;
  const float* f1cur = node1;
  float* f0nxt = f0a;
  float* f1nxt = f1a;

  const int NT = E / 16;
  const int KVGRID = (NT + WAVES_PER_BLOCK - 1) / WAVES_PER_BLOCK;
  const int EB = (E + 255) / 256;
  const int NB = (N + 255) / 256;

  for (int l = 0; l < NLAYERS; ++l) {
    fill_kernel<<<(N * 4 + 255) / 256, 256, 0, stream>>>(mbuf, -INFINITY, N * 4);
    fill_kernel<<<(N * 4 + 255) / 256, 256, 0, stream>>>(zbuf, 0.f, N * 4);
    fill_kernel<<<(N * 8 + 255) / 256, 256, 0, stream>>>(agg0, 0.f, N * 8);
    fill_kernel<<<(N * 24 + 255) / 256, 256, 0, stream>>>(agg1, 0.f, N * 24);

    q_kernel<<<NB, 256, 0, stream>>>(f0cur, f1cur, L[l].q0, L[l].q1, qf, N);

    KvArgs ka;
    ka.edge0 = edge0; ka.src = src; ka.u = ubuf; ka.scale = scale;
    ka.f0 = f0cur; ka.f1 = f1cur; ka.kv0 = kv0; ka.kv1 = kv1; ka.ntile = NT;
    for (int p = 0; p < 4; ++p) {
      const Radial& R = L[l].kv[p];
      ka.p[p] = PairP{R.w1, R.b1, R.g1, R.bb1, R.b2, R.g2, R.bb2,
                      w2f_l[l][p], w3f_l[l][p], nouts[p] / 16};
    }
    kv_kernel<<<KVGRID, TPB, 0, stream>>>(ka);

    logits_kernel<<<EB, 256, 0, stream>>>(kv0, kv1, qf, dst, wlog, mbuf, E);
    w_kernel<<<EB, 256, 0, stream>>>(scale, dst, mbuf, wlog, zbuf, E);
    agg_kernel<<<EB, 256, 0, stream>>>(wlog, zbuf, dst, kv0, kv1, agg0, agg1, E);

    node_kernel<<<NB, 256, 0, stream>>>(agg0, agg1, f0cur, f1cur, L[l].proj0, L[l].proj1,
                                        L[l].n0g, L[l].n0b, L[l].n1g, L[l].n1b,
                                        f0nxt, f1nxt, N);
    edge0_kernel<<<EB, 256, 0, stream>>>(edge0, src, f0nxt, E);

    f0cur = f0nxt; f1cur = f1nxt;
    f0nxt = (l == 0) ? f0b : f0a;
    f1nxt = (l == 0) ? f1b : f1a;
  }

  fill_kernel<<<NB, 256, 0, stream>>>((float*)d_out, 0.f, N);
  FinArgs fa;
  fa.edge0 = edge0; fa.src = src; fa.dst = dst; fa.u = ubuf; fa.scale = scale;
  fa.f0 = f0cur; fa.f1 = f1cur; fa.out = (float*)d_out; fa.ntile = NT;
  fa.p[0] = PairP{fin00.w1, fin00.b1, fin00.g1, fin00.bb1, fin00.b2, fin00.g2, fin00.bb2,
                  w2f_fin[0], w3f_fin[0], 1};
  fa.p[1] = PairP{fin10.w1, fin10.b1, fin10.g1, fin10.bb1, fin10.b2, fin10.g2, fin10.bb2,
                  w2f_fin[1], w3f_fin[1], 1};
  final_kernel<<<KVGRID, TPB, 0, stream>>>(fa);
}
